// GQAAttention_29317446762928
// MI455X (gfx1250) — matrix operand order repaired and checked by reading, not compiled, not run on hardware
//
#include <hip/hip_runtime.h>

// ---------------------------------------------------------------------------
// Types / helpers
// ---------------------------------------------------------------------------
typedef __bf16 bf16_t;
typedef float  v8f   __attribute__((ext_vector_type(8)));
typedef bf16_t v16bf __attribute__((ext_vector_type(16)));
typedef bf16_t v8bf  __attribute__((ext_vector_type(8)));
typedef bf16_t v4bf  __attribute__((ext_vector_type(4)));

union Frag {              // one WMMA 16-bit operand = 16 halves = 8 VGPRs
  v16bf v;
  struct { v8bf lo, hi; } h;
};

__device__ __forceinline__ bf16_t f2bf(float f) {
  unsigned u = __builtin_bit_cast(unsigned, f);
  u = (u + 0x7FFFu + ((u >> 16) & 1u)) >> 16;   // round-to-nearest-even
  return __builtin_bit_cast(bf16_t, (unsigned short)u);
}

// Low 32 bits of a generic (flat) shared pointer = wave-relative LDS offset.
__device__ __forceinline__ unsigned lds_off(const void* p) {
  return (unsigned)(size_t)p;
}

// CDNA5 async global->LDS copy (ASYNCcnt-tracked, bypasses VGPRs).
// Each lane copies 16 contiguous bytes: LDS[%0] = MEM[%1].
__device__ __forceinline__ void async_b128(unsigned lds, const bf16_t* g) {
  asm volatile("global_load_async_to_lds_b128 %0, %1, off"
               :: "v"(lds), "v"(g) : "memory");
}
__device__ __forceinline__ void wait_async0() {
  asm volatile("s_wait_asynccnt 0x0" ::: "memory");
}

#define SEQ     2048
#define HIDDEN  2048
#define NH      32
#define NKV     8
#define HD      64
#define NEG_BIG (-3.0e38f)

// ---------------------------------------------------------------------------
// fp32 -> bf16 straight conversion (float4 per thread)
// ---------------------------------------------------------------------------
__global__ __launch_bounds__(256) void convert_bf16(
    const float* __restrict__ src, bf16_t* __restrict__ dst)
{
  int i = (blockIdx.x * 256 + threadIdx.x) * 4;
  float4 v = *(const float4*)(src + i);
  union { v4bf v4; bf16_t e[4]; } p;
  p.e[0] = f2bf(v.x); p.e[1] = f2bf(v.y); p.e[2] = f2bf(v.z); p.e[3] = f2bf(v.w);
  *(v4bf*)(dst + i) = p.v4;
}

// ---------------------------------------------------------------------------
// fp32 [K][N] -> bf16 [N][K] transpose (LDS 32x32 tile, both sides coalesced)
// ---------------------------------------------------------------------------
__global__ __launch_bounds__(256) void transpose_bf16(
    const float* __restrict__ src, bf16_t* __restrict__ dst, int K, int N)
{
  __shared__ float tile[32][33];
  const int n0 = blockIdx.x * 32, k0 = blockIdx.y * 32;
  const int tx = threadIdx.x & 31, ty = threadIdx.x >> 5;  // ty: 0..7
  #pragma unroll
  for (int j = 0; j < 32; j += 8)
    tile[ty + j][tx] = src[(size_t)(k0 + ty + j) * N + n0 + tx];
  __syncthreads();
  #pragma unroll
  for (int j = 0; j < 32; j += 8)
    dst[(size_t)(n0 + ty + j) * K + k0 + tx] = f2bf(tile[tx][ty + j]);
}

// ---------------------------------------------------------------------------
// GEMM: C[M,N] = A[M,K] * B[K,N]; A bf16 [M][K], B pre-transposed bf16 [N][K].
// 256 thr (8 waves), tile 128x128, K-step 32, double-buffered LDS filled with
// async global->LDS B128 copies overlapped with WMMA on the other buffer.
// ---------------------------------------------------------------------------
__global__ __launch_bounds__(256) void gemm_bf16_async(
    const bf16_t* __restrict__ A, const bf16_t* __restrict__ Bt,
    float* __restrict__ C, int M, int N, int K)
{
  __shared__ alignas(16) bf16_t As[2][128][40];   // [buf][m][k] padded
  __shared__ alignas(16) bf16_t Bs[2][128][40];   // [buf][n][k] padded

  const int tid  = threadIdx.x;
  const int lane = tid & 31;
  const int wid  = tid >> 5;
  const int wm   = wid >> 1;          // 0..3 -> 32 rows each
  const int wn   = wid & 1;           // 0..1 -> 64 cols each
  const int g    = lane >> 4;
  const int ln   = lane & 15;
  const int bm   = blockIdx.y * 128;
  const int bn   = blockIdx.x * 128;

  const v8f vzero = {0.f,0.f,0.f,0.f,0.f,0.f,0.f,0.f};
  v8f acc[2][4];
  #pragma unroll
  for (int im = 0; im < 2; ++im)
    #pragma unroll
    for (int in = 0; in < 4; ++in) acc[im][in] = vzero;

  // 128 rows x 32 halves = 512 x 16B chunks per tensor; 2 chunks/thread each.
  auto issue = [&](int buf, int k0) {
    #pragma unroll
    for (int i = 0; i < 2; ++i) {
      int c = tid + i * 256;              // 0..511
      int r = c >> 2, q = (c & 3) * 8;    // 4 chunks per 32-half row
      async_b128(lds_off(&As[buf][r][q]), A  + (size_t)(bm + r) * K + k0 + q);
      async_b128(lds_off(&Bs[buf][r][q]), Bt + (size_t)(bn + r) * K + k0 + q);
    }
  };

  const int nk = K >> 5;
  issue(0, 0);
  for (int it = 0; it < nk; ++it) {
    wait_async0();        // my copies into buf(it&1) done
    __syncthreads();      // everyone's done; prior reads of buf^1 finished
    if (it + 1 < nk) issue((it + 1) & 1, (it + 1) << 5);
    const int buf = it & 1;

    Frag af[2], bf[4];
    #pragma unroll
    for (int im = 0; im < 2; ++im) {
      int m = wm * 32 + im * 16 + ln;                      // A: lane%16 = M
      af[im].h.lo = *(const v8bf*)&As[buf][m][8 * g];      // K = 8g+e
      af[im].h.hi = *(const v8bf*)&As[buf][m][16 + 8 * g]; // K = 16+8g+(e-8)
    }
    #pragma unroll
    for (int in = 0; in < 4; ++in) {
      int n = wn * 64 + in * 16 + ln;                      // B: lane%16 = N
      bf[in].h.lo = *(const v8bf*)&Bs[buf][n][8 * g];      // K = 8g+e
      bf[in].h.hi = *(const v8bf*)&Bs[buf][n][16 + 8 * g];
    }
    #pragma unroll
    for (int im = 0; im < 2; ++im)
      #pragma unroll
      for (int in = 0; in < 4; ++in)
        acc[im][in] = __builtin_amdgcn_wmma_f32_16x16x32_bf16(
            false, af[im].v, false, bf[in].v, (short)0, acc[im][in], false, false);
  }

  #pragma unroll
  for (int im = 0; im < 2; ++im)
    #pragma unroll
    for (int in = 0; in < 4; ++in)
      #pragma unroll
      for (int r = 0; r < 8; ++r) {
        int row = bm + wm * 32 + im * 16 + r + 8 * g;
        int col = bn + wn * 64 + in * 16 + ln;
        C[(size_t)row * N + col] = acc[im][in][r];
      }
}

// ---------------------------------------------------------------------------
// Per-head RMSNorm + RoPE. One wave per (token, head); lane owns dims
// {lane, lane+32} (rotate-half pair). Output head-major bf16 [head][tok][64].
// ---------------------------------------------------------------------------
__global__ __launch_bounds__(256) void norm_rope(
    const float* __restrict__ X, const float* __restrict__ lnw,
    const int* __restrict__ pos_ids, bf16_t* __restrict__ out, int nheads)
{
  const int ncols = nheads * HD;
  const int widx  = blockIdx.x * 8 + (threadIdx.x >> 5);
  const int lane  = threadIdx.x & 31;
  const int token = widx / nheads;
  const int head  = widx % nheads;

  const float* row = X + (size_t)token * ncols + head * HD;
  float x1 = row[lane], x2 = row[lane + 32];
  float ss = x1 * x1 + x2 * x2;
  #pragma unroll
  for (int off = 16; off >= 1; off >>= 1) ss += __shfl_xor(ss, off, 32);
  const float inv = rsqrtf(ss * (1.0f / 64.0f) + 1e-6f);
  x1 *= inv * lnw[lane];
  x2 *= inv * lnw[lane + 32];

  const float p = (float)pos_ids[token];
  const float ang = p * __powf(10000.0f, -(float)(2 * lane) * (1.0f / 64.0f));
  float sn, cs;
  __sincosf(ang, &sn, &cs);

  bf16_t* orow = out + ((size_t)head * SEQ + token) * HD;
  orow[lane]      = f2bf(x1 * cs - x2 * sn);
  orow[lane + 32] = f2bf(x2 * cs + x1 * sn);
}

// ---------------------------------------------------------------------------
// V: [token][kvh*64+d] f32 -> transposed bf16 [kvh][d][token] (B layout for PV)
// ---------------------------------------------------------------------------
__global__ __launch_bounds__(256) void v_relayout_t(
    const float* __restrict__ V, bf16_t* __restrict__ out)
{
  int e = blockIdx.x * 256 + threadIdx.x;  // 0 .. 2048*512
  int token = e >> 9;
  int rem = e & 511;
  int kvh = rem >> 6, d = rem & 63;
  out[(size_t)(kvh * HD + d) * SEQ + token] = f2bf(V[e]);
}

// ---------------------------------------------------------------------------
// Flash attention (causal, GQA). Grid (qblock, head), block 128 (4 waves).
// K and V^T blocks staged with double-buffered async global->LDS copies.
// Per 64-key block: 8 WMMAs for QK^T, online softmax, 8 WMMAs for PV.
// Output written as bf16 [tok][NH*HD] (A operand of the wo GEMM).
// ---------------------------------------------------------------------------
__global__ __launch_bounds__(128) void attn_fwd(
    const bf16_t* __restrict__ Q,    // [NH][SEQ][HD]
    const bf16_t* __restrict__ Kh,   // [NKV][SEQ][HD]
    const bf16_t* __restrict__ Vt,   // [NKV][HD][SEQ]  (pre-transposed)
    bf16_t* __restrict__ Obf)        // [SEQ][NH*HD]
{
  __shared__ alignas(16) bf16_t Ksh[2][64][72];   // [buf][key][dim]
  __shared__ alignas(16) bf16_t Vsh[2][64][72];   // [buf][dim][key]
  __shared__ alignas(16) bf16_t Psh[4][16][72];   // per-wave P tile

  const int qb   = blockIdx.x;
  const int head = blockIdx.y;
  const int kvh  = head >> 2;
  const int tid  = threadIdx.x;
  const int lane = tid & 31;
  const int w    = tid >> 5;
  const int g    = lane >> 4;
  const int ln   = lane & 15;

  // Q fragments (2 K-chunks over head_dim), loaded once from global.
  Frag afq[2];
  const bf16_t* qrow = Q + (((size_t)head * SEQ) + qb * 64 + w * 16 + ln) * HD;
  #pragma unroll
  for (int c = 0; c < 2; ++c) {
    afq[c].h.lo = *(const v8bf*)(qrow + c * 32 + 8 * g);
    afq[c].h.hi = *(const v8bf*)(qrow + c * 32 + 16 + 8 * g);
  }

  const v8f vzero = {0.f,0.f,0.f,0.f,0.f,0.f,0.f,0.f};
  float mrow[8], lrow[8];
  v8f o[4];
  #pragma unroll
  for (int r = 0; r < 8; ++r) { mrow[r] = NEG_BIG; lrow[r] = 0.f; }
  #pragma unroll
  for (int t = 0; t < 4; ++t) o[t] = vzero;

  const bf16_t* Kbase = Kh + (size_t)kvh * SEQ * HD;
  const bf16_t* Vbase = Vt + (size_t)kvh * HD * SEQ;

  // 64 rows x 64 halves = 512 x 16B chunks per tensor; 4 chunks/thread each.
  auto issue_kv = [&](int buf, int kb) {
    #pragma unroll
    for (int i = 0; i < 4; ++i) {
      int c = tid + i * 128;
      int r = c >> 3, q = (c & 7) * 8;
      async_b128(lds_off(&Ksh[buf][r][q]), Kbase + (size_t)(kb * 64 + r) * HD + q);
      async_b128(lds_off(&Vsh[buf][r][q]), Vbase + (size_t)r * SEQ + kb * 64 + q);
    }
  };

  issue_kv(0, 0);
  for (int kb = 0; kb <= qb; ++kb) {
    wait_async0();
    __syncthreads();
    if (kb + 1 <= qb) issue_kv((kb + 1) & 1, kb + 1);
    const int buf = kb & 1;

    // S = Q K^T over 4 key tiles.
    v8f s[4];
    #pragma unroll
    for (int t = 0; t < 4; ++t) {
      v8f cz = vzero;
      #pragma unroll
      for (int c = 0; c < 2; ++c) {
        Frag bf;
        bf.h.lo = *(const v8bf*)&Ksh[buf][t * 16 + ln][c * 32 + 8 * g];
        bf.h.hi = *(const v8bf*)&Ksh[buf][t * 16 + ln][c * 32 + 16 + 8 * g];
        cz = __builtin_amdgcn_wmma_f32_16x16x32_bf16(
            false, afq[c].v, false, bf.v, (short)0, cz, false, false);
      }
      s[t] = cz;
    }

    // Online softmax per row r (+8g).
    const int qrow0 = qb * 64 + w * 16;
    #pragma unroll
    for (int r = 0; r < 8; ++r) {
      const int mi = r + 8 * g;
      const int qpos = qrow0 + mi;
      float mx = NEG_BIG;
      #pragma unroll
      for (int t = 0; t < 4; ++t) {
        float x = s[t][r] * 0.125f;                  // HEAD_DIM^-0.5
        if (kb == qb) {
          int key = kb * 64 + t * 16 + ln;
          if (key > qpos) x = NEG_BIG;               // causal mask
        }
        s[t][r] = x;
        mx = fmaxf(mx, x);
      }
      #pragma unroll
      for (int off = 1; off < 16; off <<= 1) mx = fmaxf(mx, __shfl_xor(mx, off, 16));
      const float mnew  = fmaxf(mrow[r], mx);
      const float alpha = __expf(mrow[r] - mnew);
      mrow[r] = mnew;
      float psum = 0.f;
      #pragma unroll
      for (int t = 0; t < 4; ++t) {
        float p = __expf(s[t][r] - mnew);
        s[t][r] = p;
        psum += p;
      }
      #pragma unroll
      for (int off = 1; off < 16; off <<= 1) psum += __shfl_xor(psum, off, 16);
      lrow[r] = lrow[r] * alpha + psum;
      #pragma unroll
      for (int t = 0; t < 4; ++t) o[t][r] *= alpha;
      #pragma unroll
      for (int t = 0; t < 4; ++t) Psh[w][mi][t * 16 + ln] = f2bf(s[t][r]);
    }

    // O += P V (wave-private Psh; in-order LDS within wave).
    #pragma unroll
    for (int c = 0; c < 2; ++c) {
      Frag ap;
      ap.h.lo = *(const v8bf*)&Psh[w][ln][c * 32 + 8 * g];
      ap.h.hi = *(const v8bf*)&Psh[w][ln][c * 32 + 16 + 8 * g];
      #pragma unroll
      for (int t = 0; t < 4; ++t) {
        Frag bv;
        bv.h.lo = *(const v8bf*)&Vsh[buf][t * 16 + ln][c * 32 + 8 * g];
        bv.h.hi = *(const v8bf*)&Vsh[buf][t * 16 + ln][c * 32 + 16 + 8 * g];
        o[t] = __builtin_amdgcn_wmma_f32_16x16x32_bf16(
            false, ap.v, false, bv.v, (short)0, o[t], false, false);
      }
    }
  }

  // Normalize; write bf16 [token][head*64+dim].
  #pragma unroll
  for (int r = 0; r < 8; ++r) {
    const int tok = qb * 64 + w * 16 + r + 8 * g;
    const float inv = 1.0f / lrow[r];
    #pragma unroll
    for (int t = 0; t < 4; ++t)
      Obf[(size_t)tok * (NH * HD) + head * HD + t * 16 + ln] = f2bf(o[t][r] * inv);
  }
}

// ---------------------------------------------------------------------------
// Launch
// ---------------------------------------------------------------------------
extern "C" void kernel_launch(void* const* d_in, const int* in_sizes, int n_in,
                              void* d_out, int out_size, void* d_ws, size_t ws_size,
                              hipStream_t stream) {
  (void)in_sizes; (void)n_in; (void)out_size; (void)ws_size;
  const float* hidden = (const float*)d_in[0];
  const int*   pos    = (const int*)d_in[1];
  const float* wq     = (const float*)d_in[2];
  const float* wk     = (const float*)d_in[3];
  const float* wv     = (const float*)d_in[4];
  const float* wo     = (const float*)d_in[5];
  const float* qlnw   = (const float*)d_in[6];
  const float* klnw   = (const float*)d_in[7];

  char* ws = (char*)d_ws;
  const size_t MB = 1024ull * 1024ull;
  float*  qf    = (float*)(ws);               // 2048x2048 f32   (16 MB)
  float*  kf    = (float*)(ws + 16 * MB);     // 2048x512  f32   ( 4 MB)
  float*  vf    = (float*)(ws + 20 * MB);     // 2048x512  f32   ( 4 MB)
  bf16_t* hbf   = (bf16_t*)(ws + 24 * MB);    // hidden bf16     ( 8 MB)
  bf16_t* wq_t  = (bf16_t*)(ws + 32 * MB);    // [2048][2048]    ( 8 MB)
  bf16_t* wk_t  = (bf16_t*)(ws + 40 * MB);    // [512][2048]     ( 2 MB)
  bf16_t* wv_t  = (bf16_t*)(ws + 42 * MB);    // [512][2048]     ( 2 MB)
  bf16_t* wo_t  = (bf16_t*)(ws + 44 * MB);    // [2048][2048]    ( 8 MB)
  bf16_t* qbb   = (bf16_t*)(ws + 52 * MB);    // [32][2048][64]  ( 8 MB)
  bf16_t* kbb   = (bf16_t*)(ws + 60 * MB);    // [8][2048][64]   ( 2 MB)
  bf16_t* vtb   = (bf16_t*)(ws + 62 * MB);    // [8][64][2048]   ( 2 MB)
  bf16_t* abf   = (bf16_t*)(ws + 64 * MB);    // [2048][2048]    ( 8 MB)

  // Precision/layout conversion passes (one-time, L2-resident).
  convert_bf16<<<(SEQ * HIDDEN) / 1024, 256, 0, stream>>>(hidden, hbf);
  transpose_bf16<<<dim3(64, 64), 256, 0, stream>>>(wq, wq_t, HIDDEN, NH * HD);
  transpose_bf16<<<dim3(16, 64), 256, 0, stream>>>(wk, wk_t, HIDDEN, NKV * HD);
  transpose_bf16<<<dim3(16, 64), 256, 0, stream>>>(wv, wv_t, HIDDEN, NKV * HD);
  transpose_bf16<<<dim3(64, 64), 256, 0, stream>>>(wo, wo_t, NH * HD, HIDDEN);

  // QKV projections (async-staged bf16 WMMA, fp32 out).
  gemm_bf16_async<<<dim3(16, 16), 256, 0, stream>>>(hbf, wq_t, qf, SEQ, NH * HD, HIDDEN);
  gemm_bf16_async<<<dim3(4, 16),  256, 0, stream>>>(hbf, wk_t, kf, SEQ, NKV * HD, HIDDEN);
  gemm_bf16_async<<<dim3(4, 16),  256, 0, stream>>>(hbf, wv_t, vf, SEQ, NKV * HD, HIDDEN);

  norm_rope<<<(SEQ * NH) / 8, 256, 0, stream>>>(qf, qlnw, pos, qbb, NH);
  norm_rope<<<(SEQ * NKV) / 8, 256, 0, stream>>>(kf, klnw, pos, kbb, NKV);
  v_relayout_t<<<(SEQ * NKV * HD) / 256, 256, 0, stream>>>(vf, vtb);

  attn_fwd<<<dim3(SEQ / 64, NH), 128, 0, stream>>>(qbb, kbb, vtb, abf);

  // Output projection straight to d_out (fp32).
  gemm_bf16_async<<<dim3(16, 16), 256, 0, stream>>>(abf, wo_t, (float*)d_out,
                                                    SEQ, HIDDEN, NH * HD);
}
